// GlobalAttentionPooling_55808805044795
// MI455X (gfx1250) — compile-verified, matrix operand order checked by reading, not run on hardware
//
#include <hip/hip_runtime.h>
#include <hip/hip_bf16.h>

// ---- WMMA types (CDNA5 / gfx1250, wave32) ----
typedef __attribute__((ext_vector_type(16))) __bf16 v16bf;
typedef __attribute__((ext_vector_type(2)))  __bf16 bf16x2;
typedef __attribute__((ext_vector_type(8)))  float  v8f;

#define HDIM 128

// ---- fp32 -> packed bf16x2 conversion: prefer HW v_cvt_pk_bf16_f32 ----
__device__ __forceinline__ unsigned pack_bf16x2(float lo, float hi) {
#if __has_builtin(__builtin_amdgcn_cvt_pk_bf16_f32)
    union { bf16x2 v; unsigned u; } t;
    t.v = __builtin_amdgcn_cvt_pk_bf16_f32(lo, hi);
    return t.u;
#else
    union { bf16x2 v; unsigned u; } t;
    t.v[0] = (__bf16)lo;
    t.v[1] = (__bf16)hi;
    return t.u;
#endif
}

// ---- tanh: prefer gfx1250 V_TANH_F32 trans op ----
__device__ __forceinline__ float fast_tanh(float x) {
#if __has_builtin(__builtin_amdgcn_tanhf)
    return __builtin_amdgcn_tanhf(x);
#elif __has_builtin(__builtin_amdgcn_tanh_f32)
    return __builtin_amdgcn_tanh_f32(x);
#else
    return tanhf(x);
#endif
}

union Frag16 { v16bf v; uint4 q[2]; unsigned u[8]; };

// ---------------------------------------------------------------------------
// Kernel 1: s[n] = tanh(x[n,:] @ W1 + b1) @ w2 + b2   via v_wmma_f32_16x16x32_bf16
// Persistent blocks: W1 staged to LDS once, waves grid-stride over 16-row tiles.
// ---------------------------------------------------------------------------
__global__ __launch_bounds__(256)
void gap_score_wmma(const float* __restrict__ x,
                    const float* __restrict__ w1,
                    const float* __restrict__ b1,
                    const float* __restrict__ w2,
                    const float* __restrict__ b2,
                    float* __restrict__ s_out,
                    int N)
{
    // W1 staged transposed [n][k] in bf16 so B fragments are contiguous 16B runs.
    __shared__ __align__(16) __bf16 w1t[HDIM * HDIM];  // 32 KB

    const int tid = threadIdx.x;
    #pragma unroll 4
    for (int i = 0; i < 64; ++i) {
        int idx = tid * 64 + i;
        int n = idx >> 7, k = idx & 127;
        w1t[n * HDIM + k] = (__bf16)w1[k * HDIM + n];
    }
    __syncthreads();

    const int lane = tid & 31;
    const int wave = tid >> 5;
    const int half = lane >> 4;   // K-half selector (A & B layouts)
    const int nl   = lane & 15;

    // bias / w2 per lane column (invariant across tiles)
    float b1v[8], w2v[8];
    #pragma unroll
    for (int nt = 0; nt < 8; ++nt) {
        int n = nt * 16 + nl;
        b1v[nt] = b1[n];
        w2v[nt] = w2[n];
    }
    const float bb = b2[0];

    const int numTiles   = (N + 15) >> 4;
    const int totalWaves = gridDim.x * 8;

    for (int tile = blockIdx.x * 8 + wave; tile < numTiles; tile += totalWaves) {
        const int rowBase = tile * 16;

        // A-matrix row for this lane (lanes 0-15 and 16-31 both hold M=0..15)
        int m = rowBase + nl;
        if (m > N - 1) m = N - 1;          // clamp: keep EXEC all-ones for WMMA
        const float* __restrict__ xrow = x + (size_t)m * HDIM;

        v8f acc[8];
        const v8f zero = {0.f, 0.f, 0.f, 0.f, 0.f, 0.f, 0.f, 0.f};
        #pragma unroll
        for (int t = 0; t < 8; ++t) acc[t] = zero;

        #pragma unroll
        for (int kc = 0; kc < 4; ++kc) {
            const int kA = kc * 32 + half * 8;       // run1: K = kA..kA+7
            const int kB = kA + 16;                  // run2: K = kA+16..kA+23

            // A fragment: 16 fp32 -> 16 bf16 per lane (HW packed converts)
            Frag16 a;
            float4 f0 = *(const float4*)(xrow + kA);
            float4 f1 = *(const float4*)(xrow + kA + 4);
            float4 f2 = *(const float4*)(xrow + kB);
            float4 f3 = *(const float4*)(xrow + kB + 4);
            a.u[0] = pack_bf16x2(f0.x, f0.y);
            a.u[1] = pack_bf16x2(f0.z, f0.w);
            a.u[2] = pack_bf16x2(f1.x, f1.y);
            a.u[3] = pack_bf16x2(f1.z, f1.w);
            a.u[4] = pack_bf16x2(f2.x, f2.y);
            a.u[5] = pack_bf16x2(f2.z, f2.w);
            a.u[6] = pack_bf16x2(f3.x, f3.y);
            a.u[7] = pack_bf16x2(f3.z, f3.w);

            #pragma unroll
            for (int nt = 0; nt < 8; ++nt) {
                // B fragment: column n = nt*16+nl, two contiguous 16B runs from LDS
                Frag16 b;
                const __bf16* bp = &w1t[(nt * 16 + nl) * HDIM + kA];
                b.q[0] = *(const uint4*)bp;
                b.q[1] = *(const uint4*)(bp + 16);
                acc[nt] = __builtin_amdgcn_wmma_f32_16x16x32_bf16(
                    false, a.v, false, b.v, (short)0, acc[nt], false, false);
            }
        }

        // Epilogue: h = tanh(acc + b1), partial s = sum over this lane's 8 cols
        float part[8];
        #pragma unroll
        for (int j = 0; j < 8; ++j) part[j] = 0.f;
        #pragma unroll
        for (int nt = 0; nt < 8; ++nt) {
            #pragma unroll
            for (int j = 0; j < 8; ++j)
                part[j] += fast_tanh(acc[nt][j] + b1v[nt]) * w2v[nt];
        }

        // C layout: VGPR j -> (lanes 0-15: M=j) (lanes 16-31: M=8+j).
        // Reduce over 16 lanes of each half (xor masks 1,2,4,8 stay in-half).
        #pragma unroll
        for (int j = 0; j < 8; ++j) {
            float v = part[j];
            v += __shfl_xor(v, 1, 32);
            v += __shfl_xor(v, 2, 32);
            v += __shfl_xor(v, 4, 32);
            v += __shfl_xor(v, 8, 32);
            part[j] = v;
        }
        if (nl == 0) {  // lane 0 -> rows M=0..7, lane 16 -> rows M=8..15
            #pragma unroll
            for (int j = 0; j < 8; ++j) {
                int row = rowBase + half * 8 + j;
                if (row < N) s_out[row] = part[j] + bb;
            }
        }
    }
}

// ---------------------------------------------------------------------------
// Kernel 2: per-graph stable softmax + weighted pooling. One block per graph.
// batch is sorted -> binary-search segment bounds; no atomics.
// ---------------------------------------------------------------------------
__global__ __launch_bounds__(128)
void gap_pool(const float* __restrict__ x,
              const int* __restrict__ batch,
              const float* __restrict__ s,
              float* __restrict__ pooled,
              int N)
{
    __shared__ float red[128];
    const int g = blockIdx.x;
    const int tid = threadIdx.x;

    // lower_bound(batch, g) and lower_bound(batch, g+1)
    int lo = 0, hi = N;
    while (lo < hi) { int mid = (lo + hi) >> 1; if (batch[mid] < g) lo = mid + 1; else hi = mid; }
    const int start = lo;
    hi = N;
    while (lo < hi) { int mid = (lo + hi) >> 1; if (batch[mid] < g + 1) lo = mid + 1; else hi = mid; }
    const int end = lo;

    if (start >= end) {               // empty graph -> zeros (matches reference guards)
        pooled[(size_t)g * HDIM + tid] = 0.f;
        return;
    }

    // pass 1: segment max
    float m = -INFINITY;
    for (int n = start + tid; n < end; n += 128) m = fmaxf(m, s[n]);
    red[tid] = m; __syncthreads();
    #pragma unroll
    for (int off = 64; off > 0; off >>= 1) {
        if (tid < off) red[tid] = fmaxf(red[tid], red[tid + off]);
        __syncthreads();
    }
    float smax = red[0];
    if (!(smax > -3.0e38f && smax < 3.0e38f)) smax = 0.f;  // finite guard
    __syncthreads();

    // pass 2: sum of exp
    float sum = 0.f;
    for (int n = start + tid; n < end; n += 128) sum += __expf(s[n] - smax);
    red[tid] = sum; __syncthreads();
    #pragma unroll
    for (int off = 64; off > 0; off >>= 1) {
        if (tid < off) red[tid] += red[tid + off];
        __syncthreads();
    }
    const float denom = red[0];
    const float inv = 1.f / (denom > 0.f ? denom : 1.f);

    // pass 3: weighted feature sum; thread tid owns feature column tid (coalesced)
    float acc = 0.f;
    for (int n = start; n < end; ++n) {
        float e = __expf(s[n] - smax);
        acc += x[(size_t)n * HDIM + tid] * e;
    }
    pooled[(size_t)g * HDIM + tid] = acc * inv;
}

// ---------------------------------------------------------------------------
extern "C" void kernel_launch(void* const* d_in, const int* in_sizes, int n_in,
                              void* d_out, int out_size, void* d_ws, size_t ws_size,
                              hipStream_t stream) {
    const float* x     = (const float*)d_in[0];
    const int*   batch = (const int*)d_in[1];
    const float* w1    = (const float*)d_in[2];
    const float* b1    = (const float*)d_in[3];
    const float* w2    = (const float*)d_in[4];
    const float* b2    = (const float*)d_in[5];

    const int N = in_sizes[0] / HDIM;   // 1,000,000
    const int G = out_size   / HDIM;    // 4096

    float* s      = (float*)d_ws;       // N floats of scratch (4 MB)
    float* pooled = (float*)d_out;      // fully written by gap_pool

    const int numTiles = (N + 15) / 16;            // 62,500 wave-tiles
    int blocks = (numTiles + 7) / 8;               // upper bound
    if (blocks > 2048) blocks = 2048;              // persistent: amortize W1 staging
    gap_score_wmma<<<blocks, 256, 0, stream>>>(x, w1, b1, w2, b2, s, N);
    gap_pool<<<G, 128, 0, stream>>>(x, batch, s, pooled, N);
}
